// GNN_Explainer_24567212933212
// MI455X (gfx1250) — matrix operand order repaired and checked by reading, not compiled, not run on hardware
//
#include <hip/hip_runtime.h>
#include <stdint.h>

// Problem constants (match reference)
#define NNODES 65536
#define NEDGES 1048576
#define NGRAPH 64
#define NPER   1024
#define FEATD  64
#define HIDD   128
#define KSEL   10

typedef unsigned short u16;
typedef __attribute__((ext_vector_type(16))) __bf16 v16bf;
typedef __attribute__((ext_vector_type(8)))  float  v8f;

union FragBF { uint4 q[2]; v16bf v; };

__device__ __forceinline__ u16 f32_to_bf16(float f) {
  uint32_t u = __float_as_uint(f);
  uint32_t r = u + 0x7FFFu + ((u >> 16) & 1u);   // round-to-nearest-even
  if ((u & 0x7F800000u) == 0x7F800000u) r = u;   // inf/NaN: truncate
  return (u16)(r >> 16);
}
__device__ __forceinline__ float bf16_to_f32(u16 h) {
  return __uint_as_float(((uint32_t)h) << 16);
}

// ---------------- degree / norm ----------------
__global__ void k_deg_init(uint32_t* deg) {
  int n = blockIdx.x * blockDim.x + threadIdx.x;
  if (n < NNODES) deg[n] = 1u;                    // self-loop
}
__global__ void k_deg_count(const int* __restrict__ dst, uint32_t* deg) {
  int e = blockIdx.x * blockDim.x + threadIdx.x;
  if (e < NEDGES) atomicAdd(&deg[dst[e]], 1u);
}
__global__ void k_dinv(const uint32_t* __restrict__ deg, float* dinv) {
  int n = blockIdx.x * blockDim.x + threadIdx.x;
  if (n < NNODES) dinv[n] = rsqrtf((float)deg[n]);
}

// ---------------- conversions ----------------
__global__ void k_f32_to_bf16_vec(const float* __restrict__ in, u16* __restrict__ out, int count) {
  int t = blockIdx.x * blockDim.x + threadIdx.x;
  if (t < count) out[t] = f32_to_bf16(in[t]);
}
// W [K][Nout] f32 -> Wt [Nout][K] bf16
__global__ void k_transpose_bf16(const float* __restrict__ W, u16* __restrict__ Wt, int K, int Nout) {
  int t = blockIdx.x * blockDim.x + threadIdx.x;
  if (t < K * Nout) {
    int k = t / Nout, n = t - k * Nout;
    Wt[n * K + k] = f32_to_bf16(W[t]);
  }
}

// ---------------- WMMA GEMM (Nout fixed = HIDD = 128) ----------------
// One wave computes a full 16-row x 128-col strip: 8 accumulators, A fragment
// loaded once per K-step and reused for 8 WMMAs (8x lower A traffic).
// Optional fused epilogue: AGG[n][c] = bias[c] + C[n][c]*dinv[n]^2 (+ G2[graph][c])
__global__ void k_gemm_bf16(const u16* __restrict__ A, const u16* __restrict__ Bt,
                            float* __restrict__ C, float* __restrict__ AGG,
                            const float* __restrict__ dinv, const float* __restrict__ bias,
                            const float* __restrict__ G2, int M, int K) {
  int lane = threadIdx.x & 31;
  int wave = threadIdx.x >> 5;
  int tm = blockIdx.x * (blockDim.x >> 5) + wave;
  if (tm >= (M >> 4)) return;                      // wave-uniform exit: EXEC stays all-ones
  int half = lane >> 4;
  int l15  = lane & 15;

  const u16* arow  = A  + (size_t)(tm * 16 + l15) * K;
  const u16* bbase = Bt + (size_t)l15 * K;

  v8f acc[8];
#pragma unroll
  for (int tn = 0; tn < 8; ++tn) acc[tn] = (v8f){};

  for (int k0 = 0; k0 < K; k0 += 32) {
    FragBF fa;
    // A 16x32 bf16: elems 0..7 -> K = k0 + 8*half + j ; elems 8..15 -> +16
    const u16* ap = arow + k0 + half * 8;
    fa.q[0] = *(const uint4*)(ap);
    fa.q[1] = *(const uint4*)(ap + 16);
#pragma unroll
    for (int tn = 0; tn < 8; ++tn) {
      FragBF fb;
      // B 32x16 bf16 (stored transposed): elems 0..7 -> K = k0 + 8*half + j ; elems 8..15 -> +16
      const u16* bp = bbase + (size_t)tn * 16 * K + k0 + half * 8;
      fb.q[0] = *(const uint4*)(bp);
      fb.q[1] = *(const uint4*)(bp + 16);
      acc[tn] = __builtin_amdgcn_wmma_f32_16x16x32_bf16(false, fa.v, false, fb.v,
                                                        (short)0, acc[tn], false, false);
    }
  }

  // C/D: VGPR r -> row M = 16*tm + 8*half + r, col = 16*tn + l15
  int row0 = tm * 16 + half * 8;
  float dv2[8];
  if (bias) {
#pragma unroll
    for (int r = 0; r < 8; ++r) { float d = dinv[row0 + r]; dv2[r] = d * d; }
  }
  // all 16 rows of a tile live in the same graph (NPER % 16 == 0)
  const float* g2row = (bias && G2) ? (G2 + (size_t)(row0 >> 10) * HIDD) : nullptr;

#pragma unroll
  for (int tn = 0; tn < 8; ++tn) {
    int col = tn * 16 + l15;
    float* cp = C + (size_t)row0 * HIDD + col;
#pragma unroll
    for (int r = 0; r < 8; ++r) cp[(size_t)r * HIDD] = acc[tn][r];
    if (bias) {
      float* apg = AGG + (size_t)row0 * HIDD + col;
      float base = bias[col] + (g2row ? g2row[col] : 0.0f);
#pragma unroll
      for (int r = 0; r < 8; ++r) apg[(size_t)r * HIDD] = base + acc[tn][r] * dv2[r];
    }
  }
}

// ---------------- GCN aggregation ----------------
// per (edge, 4-feature-chunk): AGG[dst] += H[src]*norm
__global__ void k_agg_edges(const int* __restrict__ src, const int* __restrict__ dst,
                            const float* __restrict__ dinv, const float* __restrict__ H,
                            float* __restrict__ AGG) {
  long long t = (long long)blockIdx.x * blockDim.x + threadIdx.x;
  if (t >= (long long)NEDGES * (HIDD / 4)) return;
  int e  = (int)(t >> 5);              // HIDD/4 == 32
  int c4 = ((int)t & 31) * 4;
  int s = src[e], d = dst[e];
  float nm = dinv[s] * dinv[d];
  float4 hv = *(const float4*)(H + (size_t)s * HIDD + c4);
  float* ap = AGG + (size_t)d * HIDD + c4;
  atomicAdd(ap + 0, hv.x * nm);
  atomicAdd(ap + 1, hv.y * nm);
  atomicAdd(ap + 2, hv.z * nm);
  atomicAdd(ap + 3, hv.w * nm);
}

// relu in place + bf16 copy for next GEMM
__global__ void k_relu_bf16(float* __restrict__ AGG, u16* __restrict__ hbf, int count) {
  int t = blockIdx.x * blockDim.x + threadIdx.x;
  if (t >= count) return;
  float v = fmaxf(AGG[t], 0.0f);
  AGG[t] = v;
  hbf[t] = f32_to_bf16(v);
}

// ---------------- pooling / global feature ----------------
__global__ void k_pool_max(const u16* __restrict__ hbf, float* __restrict__ gpre) {
  int b = blockIdx.x, c = threadIdx.x;           // blockDim = HIDD
  float m = -3.4e38f;
  const u16* p = hbf + (size_t)b * NPER * HIDD + c;
  for (int r = 0; r < NPER; ++r) m = fmaxf(m, bf16_to_f32(p[(size_t)r * HIDD]));
  gpre[b * HIDD + c] = m;
}
__global__ void k_bias_convert(float* __restrict__ G, const float* __restrict__ bias,
                               u16* __restrict__ Gbf, int rows) {
  int t = blockIdx.x * blockDim.x + threadIdx.x;
  if (t >= rows * HIDD) return;
  int c = t & (HIDD - 1);
  float v = G[t] + bias[c];
  G[t] = v;
  Gbf[t] = f32_to_bf16(v);
}

// ---------------- W3 GEMV + logits ----------------
__global__ void k_gemv_w3(const float* __restrict__ h, const float* __restrict__ w3,
                          float* __restrict__ H3) {
  int lane = threadIdx.x & 31;
  int n = (blockIdx.x * blockDim.x + threadIdx.x) >> 5;
  if (n >= NNODES) return;
  const float* hp = h + (size_t)n * HIDD;
  float s = 0.f;
  for (int j = lane; j < HIDD; j += 32) s += hp[j] * w3[j];
  for (int off = 16; off > 0; off >>= 1) s += __shfl_xor(s, off, 32);
  if (lane == 0) H3[n] = s;
}
__global__ void k_logit_init(const float* __restrict__ H3, const float* __restrict__ dinv,
                             const float* __restrict__ b3, float* __restrict__ lg) {
  int n = blockIdx.x * blockDim.x + threadIdx.x;
  if (n >= NNODES) return;
  float di = dinv[n];
  lg[n] = b3[0] + H3[n] * di * di;
}
__global__ void k_logit_edges(const int* __restrict__ src, const int* __restrict__ dst,
                              const float* __restrict__ dinv, const float* __restrict__ H3,
                              float* __restrict__ lg) {
  int e = blockIdx.x * blockDim.x + threadIdx.x;
  if (e >= NEDGES) return;
  int s = src[e], d = dst[e];
  atomicAdd(&lg[d], H3[s] * dinv[s] * dinv[d]);
}

// ---------------- per-graph top-k threshold + mask ----------------
__global__ void k_topk_thresh(const float* __restrict__ lg, float* __restrict__ th) {
  __shared__ float vals[NPER];
  __shared__ float sv[256];
  __shared__ int   si[256];
  int b = blockIdx.x, t = threadIdx.x;           // blockDim = 256
  for (int i = t; i < NPER; i += 256) vals[i] = lg[b * NPER + i];
  __syncthreads();
  for (int it = 0; it < KSEL; ++it) {
    float bm = -3.4e38f; int bi = 0;
    for (int i = t; i < NPER; i += 256) {
      float v = vals[i];
      if (v > bm) { bm = v; bi = i; }
    }
    sv[t] = bm; si[t] = bi;
    __syncthreads();
    for (int s = 128; s > 0; s >>= 1) {
      if (t < s) {
        if (sv[t + s] > sv[t] || (sv[t + s] == sv[t] && si[t + s] < si[t])) {
          sv[t] = sv[t + s]; si[t] = si[t + s];
        }
      }
      __syncthreads();
    }
    if (t == 0) {
      if (it == KSEL - 1) th[b] = sv[0];
      vals[si[0]] = -3.4e38f;
    }
    __syncthreads();
  }
}
__global__ void k_mask(const float* __restrict__ lg, const float* __restrict__ th,
                       float* __restrict__ out) {
  int n = blockIdx.x * blockDim.x + threadIdx.x;
  if (n >= NNODES) return;
  out[n] = (lg[n] >= th[n >> 10]) ? 1.0f : 0.0f;
}

// ---------------- launcher ----------------
static inline int cdiv(long long a, int b) { return (int)((a + b - 1) / b); }

extern "C" void kernel_launch(void* const* d_in, const int* in_sizes, int n_in,
                              void* d_out, int out_size, void* d_ws, size_t ws_size,
                              hipStream_t stream) {
  (void)in_sizes; (void)n_in; (void)out_size; (void)ws_size;
  const float* x        = (const float*)d_in[0];
  const int*   esrc     = (const int*)  d_in[1];
  const int*   edst     = (const int*)  d_in[2];
  /* uniforms d_in[3] unused in eval path */
  const float* W0 = (const float*)d_in[4];  const float* b0 = (const float*)d_in[5];
  const float* W1 = (const float*)d_in[6];  const float* b1 = (const float*)d_in[7];
  const float* Wf = (const float*)d_in[8];  const float* bf = (const float*)d_in[9];
  const float* W2 = (const float*)d_in[10]; const float* b2 = (const float*)d_in[11];
  const float* W3 = (const float*)d_in[12]; const float* b3 = (const float*)d_in[13];
  float* out = (float*)d_out;

  // workspace layout
  char* ws = (char*)d_ws;
  size_t off = 0;
  auto alloc = [&](size_t bytes) -> void* {
    off = (off + 255) & ~(size_t)255;
    void* p = ws + off;
    off += bytes;
    return p;
  };
  uint32_t* deg   = (uint32_t*)alloc(NNODES * 4);
  float*    dinv  = (float*)   alloc(NNODES * 4);
  u16*      xb    = (u16*)     alloc((size_t)NNODES * FEATD * 2);
  u16*      W0t   = (u16*)     alloc((size_t)HIDD * FEATD * 2);
  u16*      W1t   = (u16*)     alloc((size_t)HIDD * HIDD * 2);
  u16*      Wft   = (u16*)     alloc((size_t)HIDD * HIDD * 2);
  u16*      W2tT  = (u16*)     alloc((size_t)HIDD * HIDD * 2);   // top half of W2, transposed
  u16*      W2bT  = (u16*)     alloc((size_t)HIDD * HIDD * 2);   // bottom half of W2, transposed
  float*    Hbuf  = (float*)   alloc((size_t)NNODES * HIDD * 4);
  float*    AGG   = (float*)   alloc((size_t)NNODES * HIDD * 4);
  u16*      hbf   = (u16*)     alloc((size_t)NNODES * HIDD * 2);
  float*    gpre  = (float*)   alloc((size_t)NGRAPH * HIDD * 4);
  u16*      gprebf= (u16*)     alloc((size_t)NGRAPH * HIDD * 2);
  float*    glob  = (float*)   alloc((size_t)NGRAPH * HIDD * 4);
  u16*      globbf= (u16*)     alloc((size_t)NGRAPH * HIDD * 2);
  float*    G2    = (float*)   alloc((size_t)NGRAPH * HIDD * 4);
  float*    H3    = (float*)   alloc(NNODES * 4);
  float*    lg    = (float*)   alloc(NNODES * 4);
  float*    th    = (float*)   alloc(NGRAPH * 4);

  const int T = 256;
  const int nhid = NNODES * HIDD;

  // 1) degree + dinv
  k_deg_init <<<cdiv(NNODES, T), T, 0, stream>>>(deg);
  k_deg_count<<<cdiv(NEDGES, T), T, 0, stream>>>(edst, deg);
  k_dinv     <<<cdiv(NNODES, T), T, 0, stream>>>(deg, dinv);

  // 2) convert x and weights to bf16 (weights transposed to [Nout][K])
  k_f32_to_bf16_vec<<<cdiv((long long)NNODES * FEATD, T), T, 0, stream>>>(x, xb, NNODES * FEATD);
  k_transpose_bf16 <<<cdiv(FEATD * HIDD, T), T, 0, stream>>>(W0, W0t, FEATD, HIDD);
  k_transpose_bf16 <<<cdiv(HIDD * HIDD, T), T, 0, stream>>>(W1, W1t, HIDD, HIDD);
  k_transpose_bf16 <<<cdiv(HIDD * HIDD, T), T, 0, stream>>>(Wf, Wft, HIDD, HIDD);
  k_transpose_bf16 <<<cdiv(HIDD * HIDD, T), T, 0, stream>>>(W2, W2tT, HIDD, HIDD);
  k_transpose_bf16 <<<cdiv(HIDD * HIDD, T), T, 0, stream>>>(W2 + (size_t)HIDD * HIDD, W2bT, HIDD, HIDD);

  // GEMM grids: 8 waves/block, one 16x128 row-strip per wave
  int gridBig   = cdiv(NNODES / 16, 8);
  int gridSmall = cdiv(NGRAPH / 16, 8);

  // 3) conv0: H = x@W0, fused AGG init (b0 + self-loop); edges; relu -> hbf (= h0 bf16)
  k_gemm_bf16<<<gridBig, T, 0, stream>>>(xb, W0t, Hbuf, AGG, dinv, b0, nullptr, NNODES, FEATD);
  k_agg_edges<<<cdiv((long long)NEDGES * 32, T), T, 0, stream>>>(esrc, edst, dinv, Hbuf, AGG);
  k_relu_bf16<<<cdiv(nhid, T), T, 0, stream>>>(AGG, hbf, nhid);

  // 4) global: max-pool h0, glob = pool@Wf + bf, G2 = glob@W2_bot
  k_pool_max<<<NGRAPH, HIDD, 0, stream>>>(hbf, gpre);
  k_f32_to_bf16_vec<<<cdiv(NGRAPH * HIDD, T), T, 0, stream>>>(gpre, gprebf, NGRAPH * HIDD);
  k_gemm_bf16<<<gridSmall, T, 0, stream>>>(gprebf, Wft, glob, nullptr, nullptr, nullptr, nullptr,
                                           NGRAPH, HIDD);
  k_bias_convert<<<cdiv(NGRAPH * HIDD, T), T, 0, stream>>>(glob, bf, globbf, NGRAPH);
  k_gemm_bf16<<<gridSmall, T, 0, stream>>>(globbf, W2bT, G2, nullptr, nullptr, nullptr, nullptr,
                                           NGRAPH, HIDD);

  // 5) conv1 applied twice (same W1)
  for (int rep = 0; rep < 2; ++rep) {
    k_gemm_bf16<<<gridBig, T, 0, stream>>>(hbf, W1t, Hbuf, AGG, dinv, b1, nullptr, NNODES, HIDD);
    k_agg_edges<<<cdiv((long long)NEDGES * 32, T), T, 0, stream>>>(esrc, edst, dinv, Hbuf, AGG);
    k_relu_bf16<<<cdiv(nhid, T), T, 0, stream>>>(AGG, hbf, nhid);
  }

  // 6) conv2: concat split -> h@W2_top, per-graph G2 folded into fused init
  k_gemm_bf16<<<gridBig, T, 0, stream>>>(hbf, W2tT, Hbuf, AGG, dinv, b2, G2, NNODES, HIDD);
  k_agg_edges<<<cdiv((long long)NEDGES * 32, T), T, 0, stream>>>(esrc, edst, dinv, Hbuf, AGG);
  k_relu_bf16<<<cdiv(nhid, T), T, 0, stream>>>(AGG, hbf, nhid);

  // 7) logits: GEMV with W3, scalar GCN aggregation
  k_gemv_w3  <<<cdiv((long long)NNODES * 32, T), T, 0, stream>>>(AGG, W3, H3);
  k_logit_init <<<cdiv(NNODES, T), T, 0, stream>>>(H3, dinv, b3, lg);
  k_logit_edges<<<cdiv(NEDGES, T), T, 0, stream>>>(esrc, edst, dinv, H3, lg);

  // 8) per-graph top-k threshold + mask
  k_topk_thresh<<<NGRAPH, 256, 0, stream>>>(lg, th);
  k_mask<<<cdiv(NNODES, T), T, 0, stream>>>(lg, th, out);
}
